// LAGAT_86543591014929
// MI455X (gfx1250) — compile-verified, matrix operand order checked by reading, not run on hardware
//
#include <hip/hip_runtime.h>
#include <hip/hip_bf16.h>
#include <math.h>

// ---------------- problem constants ----------------
#define NNODES   100000
#define NEDGES   1600000
#define ETOT     (NEDGES + NNODES)   // with self-loops
#define NFEAT    256
#define HID      32
#define HEADS    4
#define HEADS2   4
#define NCLASS   40
#define H1C      (HEADS * HID)       // 128
#define H2C      (HEADS2 * NCLASS)   // 160
#define NEG_SLOPE 0.2f

typedef __attribute__((ext_vector_type(16))) __bf16 v16bf;
typedef __attribute__((ext_vector_type(8)))  float  v8f;

union ABFrag {
    struct { int4 lo, hi; } i;
    v16bf v;
};

// ---------------- small helpers ----------------
__device__ __forceinline__ float lrelu(float x) { return x > 0.f ? x : NEG_SLOPE * x; }
__device__ __forceinline__ float eluf (float x) { return x > 0.f ? x : expf(x) - 1.f; }

__device__ __forceinline__ void atomicMaxF(float* addr, float v) {
    // signed-int max for non-negative floats, unsigned-int min for negative floats
    if (v >= 0.f) atomicMax((int*)addr, __float_as_int(v));
    else          atomicMin((unsigned int*)addr, (unsigned int)__float_as_int(v));
}

__device__ __forceinline__ void edge_sd(long long e, const int* srcp, const int* dstp,
                                        int& s, int& d) {
    if (e < NEDGES) { s = srcp[e]; d = dstp[e]; }
    else            { s = (int)(e - NEDGES); d = s; }   // self-loop
}

// ---------------- utility kernels ----------------
__global__ void fill_f32(float* p, float val, long long n) {
    long long i = (long long)blockIdx.x * blockDim.x + threadIdx.x;
    long long stride = (long long)gridDim.x * blockDim.x;
    for (; i < n; i += stride) p[i] = val;
}

__global__ void cvt_f32_bf16(const float* src, __bf16* dst, long long n) {
    long long i = (long long)blockIdx.x * blockDim.x + threadIdx.x;
    long long stride = (long long)gridDim.x * blockDim.x;
    for (; i < n; i += stride) dst[i] = (__bf16)src[i];
}

// W [K x Nout] row-major -> Wt [Nout x K] row-major, bf16
__global__ void transpose_cvt(const float* W, __bf16* Wt, int K, int Nout) {
    long long n = (long long)K * Nout;
    long long i = (long long)blockIdx.x * blockDim.x + threadIdx.x;
    long long stride = (long long)gridDim.x * blockDim.x;
    for (; i < n; i += stride) {
        int no = (int)(i / K);
        int k  = (int)(i % K);
        Wt[(long long)no * K + k] = (__bf16)W[(long long)k * Nout + no];
    }
}

// ---------------- WMMA GEMM, register-blocked over all N-tiles ----------------
// C[M x NT*16] = A[M x KK] * Bt[NT*16 x KK]^T, KK compile-time.
// One wave owns one 16-row strip of A and ALL NT output tiles. The K loop is
// fully unrolled; every fragment address is base + compile-time immediate, so
// the loop body is pure {global_load_b128 + v_wmma} with no address VALUs
// (eliminates WMMA->VALU WAR nops and lets loads pipeline ahead of WMMAs).
template <int NT, int KK>
__global__ void __launch_bounds__(256, 1)
gemm_bf16_wmma_rb(const __bf16* __restrict__ A,
                  const __bf16* __restrict__ Bt,
                  float* __restrict__ C,
                  int Mtiles) {
    const int wave  = threadIdx.x >> 5;
    const int lane  = threadIdx.x & 31;
    const int mtile = blockIdx.x * (blockDim.x >> 5) + wave;
    if (mtile >= Mtiles) return;

    const int r16  = lane & 15;
    const int half = lane >> 4;          // K sub-offset select per 16-bit A/B layout

    const __bf16* ap = A  + (size_t)(mtile * 16 + r16) * KK + half * 8;
    const __bf16* bp = Bt + (size_t)r16 * KK + half * 8;

    v8f acc[NT] = {};
#pragma unroll
    for (int k = 0; k < KK; k += 32) {
        ABFrag a;
        a.i.lo = *(const int4*)(ap + k);          // imm offset k*2
        a.i.hi = *(const int4*)(ap + k + 16);     // imm offset k*2+32
#pragma unroll
        for (int nt = 0; nt < NT; ++nt) {
            ABFrag b;
            b.i.lo = *(const int4*)(bp + nt * 16 * KK + k);        // imm offsets
            b.i.hi = *(const int4*)(bp + nt * 16 * KK + k + 16);
            acc[nt] = __builtin_amdgcn_wmma_f32_16x16x32_bf16(
                          false, a.v, false, b.v, (short)0, acc[nt], false, false);
        }
    }

    const int Nout = NT * 16;
    float* crow = C + (size_t)(mtile * 16 + half * 8) * Nout + r16;
#pragma unroll
    for (int nt = 0; nt < NT; ++nt) {
#pragma unroll
        for (int r = 0; r < 8; ++r)
            crow[r * Nout + nt * 16] = acc[nt][r];   // imm offsets
    }
}

// ---------------- attention pieces ----------------
// alpha_s[n,h] = <h[n,h,:], a_src[h,:]> ; alpha_d likewise
__global__ void alpha_kernel(const float* __restrict__ h,
                             const float* __restrict__ a_s,
                             const float* __restrict__ a_d,
                             float* __restrict__ as, float* __restrict__ ad,
                             int H, int C) {
    long long n = (long long)NNODES * H;
    long long i = (long long)blockIdx.x * blockDim.x + threadIdx.x;
    long long stride = (long long)gridDim.x * blockDim.x;
    for (; i < n; i += stride) {
        int node = (int)(i / H);
        int hh   = (int)(i % H);
        const float* hp = h + (size_t)node * H * C + (size_t)hh * C;
        float ss = 0.f, sd = 0.f;
        for (int c = 0; c < C; ++c) {
            float v = hp[c];
            ss += v * a_s[hh * C + c];
            sd += v * a_d[hh * C + c];
        }
        as[i] = ss; ad[i] = sd;
    }
}

__global__ void edge_max_kernel(const int* srcp, const int* dstp,
                                const float* __restrict__ as, const float* __restrict__ ad,
                                float* __restrict__ emax, int H) {
    long long n = (long long)ETOT * H;
    long long i = (long long)blockIdx.x * blockDim.x + threadIdx.x;
    long long stride = (long long)gridDim.x * blockDim.x;
    for (; i < n; i += stride) {
        long long e = i / H; int hh = (int)(i % H);
        int s, d; edge_sd(e, srcp, dstp, s, d);
        float ev = lrelu(as[(long long)s * H + hh] + ad[(long long)d * H + hh]);
        atomicMaxF(&emax[(long long)d * H + hh], ev);
    }
}

__global__ void edge_sum_kernel(const int* srcp, const int* dstp,
                                const float* __restrict__ as, const float* __restrict__ ad,
                                const float* __restrict__ emax,
                                float* __restrict__ den, int H) {
    long long n = (long long)ETOT * H;
    long long i = (long long)blockIdx.x * blockDim.x + threadIdx.x;
    long long stride = (long long)gridDim.x * blockDim.x;
    for (; i < n; i += stride) {
        long long e = i / H; int hh = (int)(i % H);
        int s, d; edge_sd(e, srcp, dstp, s, d);
        long long di = (long long)d * H + hh;
        float ev = lrelu(as[(long long)s * H + hh] + ad[di]);
        atomicAdd(&den[di], expf(ev - emax[di]));
    }
}

__global__ void edge_agg_kernel(const int* srcp, const int* dstp,
                                const float* __restrict__ as, const float* __restrict__ ad,
                                const float* __restrict__ emax, const float* __restrict__ den,
                                const float* __restrict__ h, float* __restrict__ out,
                                int H, int C) {
    long long n = (long long)ETOT * H;
    long long i = (long long)blockIdx.x * blockDim.x + threadIdx.x;
    long long stride = (long long)gridDim.x * blockDim.x;
    for (; i < n; i += stride) {
        long long e = i / H; int hh = (int)(i % H);
        int s, d; edge_sd(e, srcp, dstp, s, d);
        long long di = (long long)d * H + hh;
        float ev = lrelu(as[(long long)s * H + hh] + ad[di]);
        float w  = expf(ev - emax[di]) / (den[di] + 1e-16f);
        const float* hp = h + (size_t)s * H * C + (size_t)hh * C;
        float* op = out + (size_t)d * H * C + (size_t)hh * C;
        for (int c = 0; c < C; ++c) atomicAdd(&op[c], hp[c] * w);
    }
}

// x2[n, j] = elu(out{0|1}[n, j'] + bias) as bf16  (concat of two branches)
__global__ void concat_elu_kernel(const float* __restrict__ out0, const float* __restrict__ b0,
                                  const float* __restrict__ out1, const float* __restrict__ b1,
                                  __bf16* __restrict__ x2) {
    long long n = (long long)NNODES * (2 * H1C);
    long long i = (long long)blockIdx.x * blockDim.x + threadIdx.x;
    long long stride = (long long)gridDim.x * blockDim.x;
    for (; i < n; i += stride) {
        int node = (int)(i / (2 * H1C));
        int j    = (int)(i % (2 * H1C));
        float v;
        if (j < H1C) v = out0[(long long)node * H1C + j]       + b0[j];
        else         v = out1[(long long)node * H1C + (j-H1C)] + b1[j - H1C];
        x2[i] = (__bf16)eluf(v);
    }
}

// d_out[n,c] = mean_h out2[n,h,c] + b2[c]
__global__ void head_mean_kernel(const float* __restrict__ out2, const float* __restrict__ b2,
                                 float* __restrict__ y) {
    long long n = (long long)NNODES * NCLASS;
    long long i = (long long)blockIdx.x * blockDim.x + threadIdx.x;
    long long stride = (long long)gridDim.x * blockDim.x;
    for (; i < n; i += stride) {
        int node = (int)(i / NCLASS);
        int c    = (int)(i % NCLASS);
        const float* p = out2 + (size_t)node * H2C + c;
        float acc = 0.f;
#pragma unroll
        for (int hh = 0; hh < HEADS2; ++hh) acc += p[hh * NCLASS];
        y[i] = acc * (1.f / HEADS2) + b2[c];
    }
}

// ---------------- workspace layout (bytes) ----------------
#define SZ_XB   ((size_t)NNODES * NFEAT * 2)           // 51,200,000
#define OFF_X0B ((size_t)0)
#define OFF_X1B (SZ_XB)
#define OFF_X2B OFF_X0B                                 // reuse after layer-1 GEMMs
#define OFF_W10T (2 * SZ_XB)                            // 102,400,000
#define SZ_W1T  ((size_t)H1C * NFEAT * 2)              // 65,536
#define OFF_W11T (OFF_W10T + SZ_W1T)
#define OFF_W2T  (OFF_W11T + SZ_W1T)
#define SZ_W2T  ((size_t)H2C * NFEAT * 2)              // 81,920
#define OFF_H0   (OFF_W2T + SZ_W2T)                    // 102,612,992 (256-aligned)
#define SZ_H1F  ((size_t)NNODES * H1C * 4)             // 51,200,000
#define OFF_H1   (OFF_H0 + SZ_H1F)
#define OFF_H2   OFF_H0                                 // reuse (needs 64 MB < 102.4 MB)
#define OFF_OUT0 (OFF_H1 + SZ_H1F)
#define OFF_OUT1 (OFF_OUT0 + SZ_H1F)
#define OFF_OUT2 OFF_OUT0                               // reuse
#define OFF_SMALL (OFF_OUT1 + SZ_H1F)
#define SZ_NH   ((size_t)NNODES * HEADS * 4)           // 1,600,000

extern "C" void kernel_launch(void* const* d_in, const int* in_sizes, int n_in,
                              void* d_out, int out_size, void* d_ws, size_t ws_size,
                              hipStream_t stream) {
    const float* x0      = (const float*)d_in[0];
    const float* x1      = (const float*)d_in[1];
    const int*   ei      = (const int*)  d_in[2];
    const float* W1_0    = (const float*)d_in[3];
    const float* a_s10   = (const float*)d_in[4];
    const float* a_d10   = (const float*)d_in[5];
    const float* b1_0    = (const float*)d_in[6];
    const float* W1_1    = (const float*)d_in[7];
    const float* a_s11   = (const float*)d_in[8];
    const float* a_d11   = (const float*)d_in[9];
    const float* b1_1    = (const float*)d_in[10];
    const float* W2      = (const float*)d_in[11];
    const float* a_s2    = (const float*)d_in[12];
    const float* a_d2    = (const float*)d_in[13];
    const float* b2      = (const float*)d_in[14];
    float* y = (float*)d_out;

    char* ws = (char*)d_ws;
    __bf16* x0b  = (__bf16*)(ws + OFF_X0B);
    __bf16* x1b  = (__bf16*)(ws + OFF_X1B);
    __bf16* x2b  = (__bf16*)(ws + OFF_X2B);
    __bf16* w10t = (__bf16*)(ws + OFF_W10T);
    __bf16* w11t = (__bf16*)(ws + OFF_W11T);
    __bf16* w2t  = (__bf16*)(ws + OFF_W2T);
    float* h0   = (float*)(ws + OFF_H0);
    float* h1   = (float*)(ws + OFF_H1);
    float* h2   = (float*)(ws + OFF_H2);
    float* out0 = (float*)(ws + OFF_OUT0);
    float* out1 = (float*)(ws + OFF_OUT1);
    float* out2 = (float*)(ws + OFF_OUT2);
    float* as0  = (float*)(ws + OFF_SMALL + 0 * SZ_NH);
    float* ad0  = (float*)(ws + OFF_SMALL + 1 * SZ_NH);
    float* as1  = (float*)(ws + OFF_SMALL + 2 * SZ_NH);
    float* ad1  = (float*)(ws + OFF_SMALL + 3 * SZ_NH);
    float* em0  = (float*)(ws + OFF_SMALL + 4 * SZ_NH);
    float* dn0  = (float*)(ws + OFF_SMALL + 5 * SZ_NH);
    float* em1  = (float*)(ws + OFF_SMALL + 6 * SZ_NH);
    float* dn1  = (float*)(ws + OFF_SMALL + 7 * SZ_NH);
    float* as2  = (float*)(ws + OFF_SMALL + 8 * SZ_NH);
    float* ad2  = (float*)(ws + OFF_SMALL + 9 * SZ_NH);
    float* em2  = (float*)(ws + OFF_SMALL + 10 * SZ_NH);
    float* dn2  = (float*)(ws + OFF_SMALL + 11 * SZ_NH);

    const int* srcp = ei;           // edge_index[0]
    const int* dstp = ei + NEDGES;  // edge_index[1]

    const int TB = 256;
    const int GB = 2048;   // grid-stride blocks for flat kernels
    const dim3 blk(TB);

    // ---- phase 1: bf16 conversions / weight transposes ----
    cvt_f32_bf16<<<GB, blk, 0, stream>>>(x0, x0b, (long long)NNODES * NFEAT);
    cvt_f32_bf16<<<GB, blk, 0, stream>>>(x1, x1b, (long long)NNODES * NFEAT);
    transpose_cvt<<<256, blk, 0, stream>>>(W1_0, w10t, NFEAT, H1C);
    transpose_cvt<<<256, blk, 0, stream>>>(W1_1, w11t, NFEAT, H1C);
    transpose_cvt<<<256, blk, 0, stream>>>(W2,   w2t,  NFEAT, H2C);

    // ---- phase 2: layer-1 GEMMs (WMMA, register-blocked over 8 N-tiles) ----
    const int Mtiles = NNODES / 16;          // 6250
    const int WPB = TB / 32;                 // 8 waves per block
    dim3 g1((Mtiles + WPB - 1) / WPB);
    gemm_bf16_wmma_rb<H1C / 16, NFEAT><<<g1, blk, 0, stream>>>(x0b, w10t, h0, Mtiles);
    gemm_bf16_wmma_rb<H1C / 16, NFEAT><<<g1, blk, 0, stream>>>(x1b, w11t, h1, Mtiles);

    // ---- phase 3: layer-1 attention ----
    alpha_kernel<<<GB, blk, 0, stream>>>(h0, a_s10, a_d10, as0, ad0, HEADS, HID);
    alpha_kernel<<<GB, blk, 0, stream>>>(h1, a_s11, a_d11, as1, ad1, HEADS, HID);

    fill_f32<<<GB, blk, 0, stream>>>(em0, -INFINITY, (long long)NNODES * HEADS);
    fill_f32<<<GB, blk, 0, stream>>>(dn0, 0.f,       (long long)NNODES * HEADS);
    fill_f32<<<GB, blk, 0, stream>>>(em1, -INFINITY, (long long)NNODES * HEADS);
    fill_f32<<<GB, blk, 0, stream>>>(dn1, 0.f,       (long long)NNODES * HEADS);
    fill_f32<<<GB, blk, 0, stream>>>(out0, 0.f, (long long)NNODES * H1C);
    fill_f32<<<GB, blk, 0, stream>>>(out1, 0.f, (long long)NNODES * H1C);

    edge_max_kernel<<<GB, blk, 0, stream>>>(srcp, dstp, as0, ad0, em0, HEADS);
    edge_sum_kernel<<<GB, blk, 0, stream>>>(srcp, dstp, as0, ad0, em0, dn0, HEADS);
    edge_agg_kernel<<<GB, blk, 0, stream>>>(srcp, dstp, as0, ad0, em0, dn0, h0, out0, HEADS, HID);

    edge_max_kernel<<<GB, blk, 0, stream>>>(srcp, dstp, as1, ad1, em1, HEADS);
    edge_sum_kernel<<<GB, blk, 0, stream>>>(srcp, dstp, as1, ad1, em1, dn1, HEADS);
    edge_agg_kernel<<<GB, blk, 0, stream>>>(srcp, dstp, as1, ad1, em1, dn1, h1, out1, HEADS, HID);

    // ---- phase 4: concat + elu -> bf16 x2 (overwrites x0b region) ----
    concat_elu_kernel<<<GB, blk, 0, stream>>>(out0, b1_0, out1, b1_1, x2b);

    // ---- phase 5: layer-2 GEMM (WMMA, register-blocked over 10 N-tiles) ----
    gemm_bf16_wmma_rb<H2C / 16, NFEAT><<<g1, blk, 0, stream>>>(x2b, w2t, h2, Mtiles);

    // ---- phase 6: layer-2 attention ----
    alpha_kernel<<<GB, blk, 0, stream>>>(h2, a_s2, a_d2, as2, ad2, HEADS2, NCLASS);
    fill_f32<<<GB, blk, 0, stream>>>(em2, -INFINITY, (long long)NNODES * HEADS2);
    fill_f32<<<GB, blk, 0, stream>>>(dn2, 0.f,       (long long)NNODES * HEADS2);
    fill_f32<<<GB, blk, 0, stream>>>(out2, 0.f, (long long)NNODES * H2C);

    edge_max_kernel<<<GB, blk, 0, stream>>>(srcp, dstp, as2, ad2, em2, HEADS2);
    edge_sum_kernel<<<GB, blk, 0, stream>>>(srcp, dstp, as2, ad2, em2, dn2, HEADS2);
    edge_agg_kernel<<<GB, blk, 0, stream>>>(srcp, dstp, as2, ad2, em2, dn2, h2, out2, HEADS2, NCLASS);

    // ---- phase 7: head mean + bias -> output ----
    head_mean_kernel<<<GB, blk, 0, stream>>>(out2, b2, y);
}
